// Net_82489141887760
// MI455X (gfx1250) — matrix operand order repaired and checked by reading, not compiled, not run on hardware
//
#include <hip/hip_runtime.h>
#include <hip/hip_bf16.h>
#include <math.h>

typedef __attribute__((ext_vector_type(16))) _Float16 v16h;
typedef __attribute__((ext_vector_type(8)))  float    v8f;
typedef __attribute__((ext_vector_type(4)))  unsigned int v4u;

#define N_NODES  40000
#define N_EDGES  400000
#define N_GRAPHS 256
#define HEADS    32
#define EA       (N_EDGES + N_NODES)   // edges + self-loops
#define BN_EPS   1e-5f

// ---------- helpers ----------
static __device__ __forceinline__ unsigned enc_f32(float x) {
  unsigned u = __float_as_uint(x);
  return (u & 0x80000000u) ? ~u : (u | 0x80000000u);
}
static __device__ __forceinline__ float dec_f32(unsigned e) {
  return (e & 0x80000000u) ? __uint_as_float(e & 0x7fffffffu) : __uint_as_float(~e);
}
static __device__ __forceinline__ void edge_sd(const int* __restrict__ ei, int e, int& s, int& d) {
  if (e < N_EDGES) { s = ei[e]; d = ei[N_EDGES + e]; }
  else             { s = e - N_EDGES; d = s; }        // appended self-loops
}

// ---------- fills ----------
__global__ void k_fill_f32(float* p, float v, int n) {
  int i = blockIdx.x * blockDim.x + threadIdx.x;
  if (i < n) p[i] = v;
}
__global__ void k_fill_u32(unsigned* p, unsigned v, int n) {
  int i = blockIdx.x * blockDim.x + threadIdx.x;
  if (i < n) p[i] = v;
}

// ---------- BatchNorm (training stats, biased var) ----------
__global__ void k_bn_stats(const float* __restrict__ x, float* mean, float* rstd, int n, int c) {
  int col = blockIdx.x;
  __shared__ float s1[256], s2[256];
  float a = 0.f, b = 0.f;
  for (int r = threadIdx.x; r < n; r += blockDim.x) {
    float v = x[(size_t)r * c + col];
    a += v; b += v * v;
  }
  s1[threadIdx.x] = a; s2[threadIdx.x] = b;
  __syncthreads();
  for (int s = 128; s > 0; s >>= 1) {
    if ((int)threadIdx.x < s) { s1[threadIdx.x] += s1[threadIdx.x + s]; s2[threadIdx.x] += s2[threadIdx.x + s]; }
    __syncthreads();
  }
  if (threadIdx.x == 0) {
    float m   = s1[0] / (float)n;
    float var = s2[0] / (float)n - m * m;
    mean[col] = m;
    rstd[col] = rsqrtf(var + BN_EPS);
  }
}

// BN-apply + f32->f16 convert, zero-padded K dimension for WMMA
__global__ void k_bn_apply_h(const float* __restrict__ x, const float* __restrict__ mean,
                             const float* __restrict__ rstd, const float* __restrict__ g,
                             const float* __restrict__ be, _Float16* __restrict__ out,
                             int n, int cin, int kpad) {
  int i = blockIdx.x * blockDim.x + threadIdx.x;
  if (i >= n * kpad) return;
  int r = i / kpad, k = i - r * kpad;
  float v = 0.f;
  if (k < cin) v = (x[(size_t)r * cin + k] - mean[k]) * rstd[k] * g[k] + be[k];
  out[i] = (_Float16)v;
}

// W [cin, ncols] f32  ->  Wt [ncols, kpad] f16 (transposed, K zero-padded)
__global__ void k_wcvt(const float* __restrict__ W, _Float16* __restrict__ Wt,
                       int cin, int ncols, int kpad) {
  int i = blockIdx.x * blockDim.x + threadIdx.x;
  if (i >= ncols * kpad) return;
  int col = i / kpad, k = i - col * kpad;
  float v = (k < cin) ? W[(size_t)k * ncols + col] : 0.f;
  Wt[i] = (_Float16)v;
}

// ---------- WMMA GEMM: H[n, ncols] = A[n, kpad] @ Bt[ncols, kpad]^T ----------
// 128 threads = 4 waves; each wave computes one 16x16 tile with v_wmma_f32_16x16x32_f16.
__global__ void k_gemm_wmma(const _Float16* __restrict__ A, const _Float16* __restrict__ Bt,
                            _Float16* __restrict__ H, int kpad, int ncols) {
  int lane = threadIdx.x & 31;
  int wv   = threadIdx.x >> 5;
  int tileM = blockIdx.y * 4 + wv;
  int tileN = blockIdx.x;
  int half = lane >> 4;      // lane group 0-15 vs 16-31
  int lm   = lane & 15;

  const _Float16* arow = A  + (size_t)(tileM * 16 + lm) * kpad;
  const _Float16* brow = Bt + (size_t)(tileN * 16 + lm) * kpad;

  v8f c = {};
  for (int kb = 0; kb < kpad; kb += 32) {
    // A fragment per ISA 16-bit 16x32 layout:
    // element e=2p+b  ->  K = kb + (p<4?0:16) + half*8 + (p&3)*2 + b
    union { v16h v; unsigned u[8]; } af;
    const _Float16* ap = arow + kb + half * 8;
#pragma unroll
    for (int p = 0; p < 4; ++p) {
      af.u[p]     = *(const unsigned*)(ap + p * 2);        // K pairs 0..7 (+half*8)
      af.u[4 + p] = *(const unsigned*)(ap + 16 + p * 2);   // K pairs 16..23 (+half*8)
    }
    // B fragment: lane column N=lm, element e -> K = kb + (e<8?0:16) + half*8 + (e&7) (same as A)
    union { v16h v; v4u q[2]; } bf;
    const _Float16* bp = brow + kb + half * 8;
    bf.q[0] = *(const v4u*)(bp);
    bf.q[1] = *(const v4u*)(bp + 16);

    c = __builtin_amdgcn_wmma_f32_16x16x32_f16(false, af.v, false, bf.v,
                                               (short)0, c, false, false);
  }
  // D layout: VGPR r -> M = r + 8*half, N = lm ; store as f16 (halves gather traffic later)
  _Float16* hp = H + (size_t)(tileM * 16 + half * 8) * ncols + tileN * 16 + lm;
#pragma unroll
  for (int r = 0; r < 8; ++r) hp[(size_t)r * ncols] = (_Float16)c[r];
}

// ---------- attention scores per (node, head) ----------
__global__ void k_scores(const _Float16* __restrict__ h, const float* __restrict__ a_s,
                         const float* __restrict__ a_d, float* __restrict__ ss,
                         float* __restrict__ sd, int co) {
  int i = blockIdx.x * blockDim.x + threadIdx.x;
  if (i >= N_NODES * HEADS) return;
  int n = i >> 5, hh = i & 31;
  const _Float16* hp = h + (size_t)n * HEADS * co + (size_t)hh * co;
  float a = 0.f, b = 0.f;
  for (int c2 = 0; c2 < co; ++c2) {
    float v = (float)hp[c2];
    a += v * a_s[hh * co + c2];
    b += v * a_d[hh * co + c2];
  }
  ss[i] = a; sd[i] = b;
}

// ---------- edge pass 1: leaky logits + segment max (monotonic uint atomicMax) ----------
__global__ void k_edge_logit(const int* __restrict__ ei, const float* __restrict__ ss,
                             const float* __restrict__ sd, float* __restrict__ ebuf,
                             unsigned* __restrict__ menc) {
  int i = blockIdx.x * blockDim.x + threadIdx.x;
  if (i >= EA * HEADS) return;
  int e = i >> 5, hh = i & 31;
  int s, d; edge_sd(ei, e, s, d);
  float l = ss[s * HEADS + hh] + sd[d * HEADS + hh];
  l = (l > 0.f) ? l : 0.2f * l;
  ebuf[i] = l;
  atomicMax(&menc[d * HEADS + hh], enc_f32(l));
}

// ---------- edge pass 2: exp(logit - max), segment sum ----------
__global__ void k_edge_exp(const int* __restrict__ ei, float* __restrict__ ebuf,
                           const unsigned* __restrict__ menc, float* __restrict__ denom) {
  int i = blockIdx.x * blockDim.x + threadIdx.x;
  if (i >= EA * HEADS) return;
  int e = i >> 5, hh = i & 31;
  int s, d; edge_sd(ei, e, s, d);
  float m  = dec_f32(menc[d * HEADS + hh]);
  float ex = __expf(ebuf[i] - m);
  ebuf[i] = ex;
  atomicAdd(&denom[d * HEADS + hh], ex);
}

// ---------- edge pass 3: aggregation (one wave per edge) ----------
__global__ void k_edge_agg(const int* __restrict__ ei, const float* __restrict__ ebuf,
                           const float* __restrict__ denom, const _Float16* __restrict__ h,
                           float* __restrict__ out, int co) {
  __shared__ float alph[8][HEADS];
  int wv = threadIdx.x >> 5, lane = threadIdx.x & 31;
  int e = blockIdx.x * 8 + wv;
  if (e >= EA) return;
  int s, d; edge_sd(ei, e, s, d);
  // per-head alpha, one per lane, broadcast via LDS
  alph[wv][lane] = ebuf[(size_t)e * HEADS + lane] / denom[d * HEADS + lane];
  const _Float16* hs = h + (size_t)s * HEADS * co;
  __builtin_prefetch(hs, 0, 0);   // global_prefetch_b8 of gathered src row
  for (int c2 = lane; c2 < co; c2 += 32) {
    float acc = 0.f;
#pragma unroll 8
    for (int hh = 0; hh < HEADS; ++hh)
      acc += alph[wv][hh] * (float)hs[hh * co + c2];
    atomicAdd(&out[(size_t)d * co + c2], acc * (1.0f / HEADS));
  }
}

// ---------- bias + leaky_relu(0.01) ----------
__global__ void k_bias_lrelu(float* __restrict__ x, const float* __restrict__ b, int co, int n) {
  int i = blockIdx.x * blockDim.x + threadIdx.x;
  if (i >= n * co) return;
  int c2 = i % co;
  float v = x[i] + b[c2];
  x[i] = (v > 0.f) ? v : 0.01f * v;
}

// ---------- graph mean-pool ----------
__global__ void k_pool(const float* __restrict__ feat, const int* __restrict__ batch,
                       float* __restrict__ pooled, float* __restrict__ counts) {
  int n = blockIdx.x * blockDim.x + threadIdx.x;
  if (n >= N_NODES) return;
  int b = batch[n];
  atomicAdd(&counts[b], 1.f);
  for (int c = 0; c < 5; ++c)
    atomicAdd(&pooled[b * 5 + c], feat[(size_t)n * 5 + c]);
}

// ---------- final FC + sigmoid ----------
__global__ void k_fc(const float* __restrict__ pooled, const float* __restrict__ counts,
                     const float* __restrict__ cond, const float* __restrict__ fcW,
                     const float* __restrict__ fcb, float* __restrict__ out) {
  int g = blockIdx.x * blockDim.x + threadIdx.x;
  if (g >= N_GRAPHS) return;
  float cnt = fmaxf(counts[g], 1.f);
  float s = fcb[0];
  for (int c = 0; c < 5; ++c)   s += (pooled[g * 5 + c] / cnt) * fcW[c];
  for (int j = 0; j < 100; ++j) s += cond[j] * fcW[5 + j];
  out[g] = 1.f / (1.f + __expf(-s));
}

extern "C" void kernel_launch(void* const* d_in, const int* in_sizes, int n_in,
                              void* d_out, int out_size, void* d_ws, size_t ws_size,
                              hipStream_t stream) {
  (void)in_sizes; (void)n_in; (void)out_size; (void)ws_size;

  const float* x     = (const float*)d_in[0];
  const int*   ei    = (const int*)d_in[1];
  const int*   batch = (const int*)d_in[2];
  const float* cond  = (const float*)d_in[3];
  const float* W [4] = {(const float*)d_in[4],  (const float*)d_in[8],  (const float*)d_in[12], (const float*)d_in[16]};
  const float* As[4] = {(const float*)d_in[5],  (const float*)d_in[9],  (const float*)d_in[13], (const float*)d_in[17]};
  const float* Ad[4] = {(const float*)d_in[6],  (const float*)d_in[10], (const float*)d_in[14], (const float*)d_in[18]};
  const float* Bi[4] = {(const float*)d_in[7],  (const float*)d_in[11], (const float*)d_in[15], (const float*)d_in[19]};
  const float* G [4] = {(const float*)d_in[20], (const float*)d_in[22], (const float*)d_in[24], (const float*)d_in[26]};
  const float* Be[4] = {(const float*)d_in[21], (const float*)d_in[23], (const float*)d_in[25], (const float*)d_in[27]};
  const float* fcW   = (const float*)d_in[30];
  const float* fcb   = (const float*)d_in[31];

  const int Cin[4] = {128, 90, 45, 15};
  const int Co [4] = { 90, 45, 15,  5};
  const int Kp [4] = {128, 96, 64, 32};   // K padded to multiple of 32 for WMMA

  // ---- carve workspace ----
  char* base = (char*)d_ws;
  size_t off = 0;
  auto carve = [&](size_t bytes) -> void* {
    off = (off + 255) & ~(size_t)255;
    void* p = base + off;
    off += bytes;
    return p;
  };
  float*     mean   = (float*)carve(128 * sizeof(float));
  float*     rstd   = (float*)carve(128 * sizeof(float));
  _Float16*  Xh     = (_Float16*)carve((size_t)N_NODES * 128 * sizeof(_Float16));
  _Float16*  Wt     = (_Float16*)carve((size_t)2880 * 128 * sizeof(_Float16));
  _Float16*  Hbuf   = (_Float16*)carve((size_t)N_NODES * 2880 * sizeof(_Float16)); // f16 h => half gather BW
  float*     ssrc   = (float*)carve((size_t)N_NODES * HEADS * sizeof(float));
  float*     sdst   = (float*)carve((size_t)N_NODES * HEADS * sizeof(float));
  float*     denom  = (float*)carve((size_t)N_NODES * HEADS * sizeof(float));
  unsigned*  menc   = (unsigned*)carve((size_t)N_NODES * HEADS * sizeof(unsigned));
  float*     ebuf   = (float*)carve((size_t)EA * HEADS * sizeof(float));
  float*     fA     = (float*)carve((size_t)N_NODES * 90 * sizeof(float));
  float*     fB     = (float*)carve((size_t)N_NODES * 90 * sizeof(float));
  float*     pooled = (float*)carve((size_t)N_GRAPHS * 5 * sizeof(float));
  float*     counts = (float*)carve((size_t)N_GRAPHS * sizeof(float));

  const float* Xcur = x;
  float* outBuf = fA;
  float* altBuf = fB;

  for (int L = 0; L < 4; ++L) {
    const int ci = Cin[L], co = Co[L], kp = Kp[L], hc = HEADS * co;

    k_bn_stats<<<ci, 256, 0, stream>>>(Xcur, mean, rstd, N_NODES, ci);

    int na = N_NODES * kp;
    k_bn_apply_h<<<(na + 255) / 256, 256, 0, stream>>>(Xcur, mean, rstd, G[L], Be[L],
                                                       Xh, N_NODES, ci, kp);
    int nw = hc * kp;
    k_wcvt<<<(nw + 255) / 256, 256, 0, stream>>>(W[L], Wt, ci, hc, kp);

    dim3 gg(hc / 16, N_NODES / 64);          // 16-col tiles x (4 waves * 16 rows) blocks
    k_gemm_wmma<<<gg, 128, 0, stream>>>(Xh, Wt, Hbuf, kp, hc);

    k_scores<<<(N_NODES * HEADS + 255) / 256, 256, 0, stream>>>(Hbuf, As[L], Ad[L],
                                                                ssrc, sdst, co);

    k_fill_u32<<<(N_NODES * HEADS + 255) / 256, 256, 0, stream>>>(menc, 0u, N_NODES * HEADS);
    k_fill_f32<<<(N_NODES * HEADS + 255) / 256, 256, 0, stream>>>(denom, 0.f, N_NODES * HEADS);
    k_fill_f32<<<(N_NODES * co + 255) / 256, 256, 0, stream>>>(outBuf, 0.f, N_NODES * co);

    int net = EA * HEADS;
    k_edge_logit<<<(net + 255) / 256, 256, 0, stream>>>(ei, ssrc, sdst, ebuf, menc);
    k_edge_exp  <<<(net + 255) / 256, 256, 0, stream>>>(ei, ebuf, menc, denom);
    k_edge_agg  <<<(EA + 7) / 8,      256, 0, stream>>>(ei, ebuf, denom, Hbuf, outBuf, co);

    k_bias_lrelu<<<(N_NODES * co + 255) / 256, 256, 0, stream>>>(outBuf, Bi[L], co, N_NODES);

    Xcur = outBuf;
    float* t = outBuf; outBuf = altBuf; altBuf = t;
  }

  k_fill_f32<<<(N_GRAPHS * 5 + 255) / 256, 256, 0, stream>>>(pooled, 0.f, N_GRAPHS * 5);
  k_fill_f32<<<(N_GRAPHS + 255) / 256, 256, 0, stream>>>(counts, 0.f, N_GRAPHS);
  k_pool<<<(N_NODES + 255) / 256, 256, 0, stream>>>(Xcur, batch, pooled, counts);
  k_fc<<<1, 256, 0, stream>>>(pooled, counts, cond, fcW, fcb, (float*)d_out);
}
